// AutoBoxGraphAttention_72636486910241
// MI455X (gfx1250) — compile-verified, matrix operand order checked by reading, not run on hardware
//
#include <hip/hip_runtime.h>
#include <hip/hip_bf16.h>
#include <math.h>

// ---------------------------------------------------------------------------
// CDNA5 (gfx1250) WMMA implementation of AutoBoxGraphAttention
// Heavy GEMMs on v_wmma_f32_16x16x32_bf16 (bf16 in, f32 accumulate).
// Activations stored with a zero halo (66x66) so implicit-GEMM A loads are
// branch-free.  Fusion conv weights streamed through LDS with the Tensor
// Data Mover (tensor_load_to_lds + s_wait_tensorcnt), double-buffered;
// staged weights are read back with explicit addrspace(3) ds_load_b128.
// ---------------------------------------------------------------------------

typedef __attribute__((ext_vector_type(16))) __bf16 v16bf;
typedef __attribute__((ext_vector_type(8)))  float  v8f;
typedef __attribute__((ext_vector_type(4)))  unsigned int v4u;
typedef __attribute__((ext_vector_type(8)))  int v8i_t;
typedef __attribute__((ext_vector_type(4)))  int v4i_t;
typedef __attribute__((ext_vector_type(4)))  unsigned int u32x4;

#define DEV __device__ __forceinline__

#if defined(__AMDGCN__) && __has_builtin(__builtin_amdgcn_tensor_load_to_lds)
#define HAVE_TDM 1
#else
#define HAVE_TDM 0
#endif

// Device-pass-only check (host pass also defines __AMDGCN__ on this driver,
// but lacks the builtin).  Round-4 result proved the device pass has it.
#if defined(__HIP_DEVICE_COMPILE__)
static_assert(HAVE_TDM == 1, "gfx1250 device pass must have tensor_load_to_lds");
#endif

#if defined(__AMDGCN__) && __has_builtin(__builtin_amdgcn_s_wait_tensorcnt)
#define WAIT_TENSORCNT(n) __builtin_amdgcn_s_wait_tensorcnt(n)
#else
#define WAIT_TENSORCNT(n)
#endif

// padded spatial layout: [b][66][66][512]
#define PW 66
#define PPIX (PW * PW)                  // 4356 pixels per batch
#define PAD_ELEMS (8 * PPIX * 512)      // 17,842,176

union FragB { v16bf v; uint4 u[2]; u32x4 w[2]; };

// A/B fragment from a row-major (row, k) bf16 matrix (ISA 7.12.2 layout):
// row = lane&15; lanes 0-15 hold K runs {0..7},{16..23}, lanes 16-31 hold
// {8..15},{24..31} -> two contiguous 16B loads per lane.
DEV v16bf load_frag_rm(const __bf16* base, int row, int ld, int k0) {
  FragB f;
  int off = (threadIdx.x & 16) ? 8 : 0;
  const __bf16* p = base + (size_t)row * ld + k0 + off;
  f.u[0] = *(const uint4*)p;
  f.u[1] = *(const uint4*)(p + 16);
  return f.v;
}

// Fragment from pre-packed per-lane storage (16 contiguous bf16).
DEV v16bf load_frag_packed(const __bf16* p) {
  FragB f;
  f.u[0] = *(const uint4*)p;
  f.u[1] = *(const uint4*)(p + 8);
  return f.v;
}

#if HAVE_TDM
// Fragment from LDS byte offset via explicit addrspace(3) loads (ds_load_b128).
typedef __attribute__((address_space(3))) const u32x4 lds_u32x4;
DEV v16bf load_frag_lds(unsigned boff) {
  lds_u32x4* p = (lds_u32x4*)(size_t)boff;
  FragB f;
  f.w[0] = p[0];
  f.w[1] = p[1];
  return f.v;
}
#endif

DEV int frag_k(int l, int j) { return ((l & 16) ? 8 : 0) + (j < 8 ? j : j + 8); }

DEV float gelu_exact(float v) { return 0.5f * v * (1.0f + erff(v * 0.70710678118f)); }
DEV float sigmoidf_(float v)  { return 1.0f / (1.0f + expf(-v)); }

#if HAVE_TDM
// TDM: load a 2D tile (4 rows x 1024B, row stride 288KB) of packed weights
// into LDS at byte offset lds_off.  Descriptor per cdna5 ISA 8.3/8.4.
DEV void tdm_load_w(const __bf16* gsrc, unsigned lds_off) {
  unsigned long long ga = (unsigned long long)(uintptr_t)gsrc;
  unsigned lo = (unsigned)__builtin_amdgcn_readfirstlane((int)(ga & 0xffffffffu));
  unsigned hi = (unsigned)__builtin_amdgcn_readfirstlane((int)(ga >> 32));
  unsigned lx = (unsigned)__builtin_amdgcn_readfirstlane((int)lds_off);
  v4u g0;
  g0.x = 1u;                                  // count=1, user descriptor
  g0.y = lx;                                  // lds_addr (bytes)
  g0.z = lo;                                  // global_addr[31:0]
  g0.w = (hi & 0x1ffffffu) | (2u << 30);      // global_addr[56:32] | type=2
  v8i_t g1;
  g1[0] = (2 << 16);                          // data_size = 4B, no mask/pad
  g1[1] = 0;                                  // tensor_dim0[15:0] of 2^30 = 0
  g1[2] = 0x4000 | (4 << 16);                 // tensor_dim0 hi, tensor_dim1 = 4
  g1[3] = (256 << 16);                        // tile_dim0 = 256 dwords (1KB)
  g1[4] = 4;                                  // tile_dim1 = 4 rows
  g1[5] = 73728;                              // tensor_dim0_stride = 288KB/4
  g1[6] = 0;
  g1[7] = 0;
  v4i_t z = {0, 0, 0, 0};
#if __clang_major__ >= 23
  v8i_t z8 = {0, 0, 0, 0, 0, 0, 0, 0};
  __builtin_amdgcn_tensor_load_to_lds(g0, g1, z, z, z8, 0);
#else
  __builtin_amdgcn_tensor_load_to_lds(g0, g1, z, z, 0);
#endif
}
#endif

// ---------------------------------------------------------------------------
// Prep kernels
// ---------------------------------------------------------------------------

// x: NCHW f32 -> xh: padded NHWC bf16 with zero halo
__global__ void k_cast_x_pad(const float* __restrict__ x, __bf16* __restrict__ xh) {
  int tid = blockIdx.x * 256 + threadIdx.x;            // PAD_ELEMS
  int c = tid & 511;
  int t = tid >> 9;
  int xx = t % PW;
  int t2 = t / PW;
  int yy = t2 % PW;
  int b  = t2 / PW;
  float v = 0.f;
  if (yy >= 1 && yy <= 64 && xx >= 1 && xx <= 64)
    v = x[(((b * 512 + c) * 64 + (yy - 1)) * 64) + (xx - 1)];
  xh[tid] = (__bf16)v;
}

__global__ void k_zero_bf16(__bf16* __restrict__ p) {
  int tid = blockIdx.x * 256 + threadIdx.x;
  p[tid] = (__bf16)0.f;
}

// Pack OIHW f32 weights into per-lane WMMA B-fragment order (bf16).
// s-major:  pk[((s*(N/16)+n0)*32+l)*16+j];  nt-major: pk[((n0*Ks+s)*32+l)*16+j]
__global__ void k_pack_w(const float* __restrict__ w, __bf16* __restrict__ pk,
                         int K, int N, int Cin, int is3x3, int nt_major) {
  int idx = blockIdx.x * 256 + threadIdx.x;
  if (idx >= K * N) return;
  int j = idx & 15;
  int l = (idx >> 4) & 31;
  int t = idx >> 9;
  int ntiles = N >> 4, ksteps = K >> 5;
  int n0, s;
  if (nt_major) { n0 = t / ksteps; s = t % ksteps; }
  else          { n0 = t % ntiles; s = t / ntiles; }
  int k = s * 32 + frag_k(l, j);
  int n = n0 * 16 + (l & 15);
  int c = k % Cin;
  int tap = k / Cin;
  int si;
  if (is3x3) { int kh = tap / 3, kw = tap % 3; si = ((n * Cin + c) * 3 + kh) * 3 + kw; }
  else       { si = n * Cin + c; }
  pk[idx] = (__bf16)w[si];
}

// ---------------------------------------------------------------------------
// box_net conv3x3 512->64 + bias + GELU (implicit GEMM, WMMA)
// M=32768, N=64, K=4608.  Block = 4 waves; wave: 16 M rows x all 4 N-tiles.
// ---------------------------------------------------------------------------
__global__ void k_box_conv1(const __bf16* __restrict__ xh, const __bf16* __restrict__ pw,
                            const float* __restrict__ bias, __bf16* __restrict__ t1) {
  int wave = threadIdx.x >> 5, lane = threadIdx.x & 31;
  int m0 = blockIdx.x * 64 + wave * 16;
  int row = lane & 15;
  int m = m0 + row;
  int b = m >> 12, y = (m >> 6) & 63, xw = m & 63;
  v8f acc[4] = {};
  for (int tap = 0; tap < 9; ++tap) {
    int pix = (b * PW + y + tap / 3) * PW + (xw + tap % 3);  // halo handles bounds
    for (int cs = 0; cs < 16; ++cs) {                  // 512 cin / 32
      v16bf a = load_frag_rm(xh, pix, 512, cs * 32);
      int s = tap * 16 + cs;
#pragma unroll
      for (int nt = 0; nt < 4; ++nt) {
        v16bf bf = load_frag_packed(pw + ((size_t)((s * 4 + nt) * 32 + lane) << 4));
        acc[nt] = __builtin_amdgcn_wmma_f32_16x16x32_bf16(false, a, false, bf,
                                                          (short)0, acc[nt], false, false);
      }
    }
  }
  int col = lane & 15;
  int rb = (lane & 16) ? 8 : 0;
#pragma unroll
  for (int nt = 0; nt < 4; ++nt)
    for (int r = 0; r < 8; ++r) {
      int mr = m0 + rb + r;
      int n = nt * 16 + col;
      t1[(size_t)mr * 64 + n] = (__bf16)gelu_exact(acc[nt][r] + bias[n]);
    }
}

// box_net conv1x1 64->4 + sigmoid -> boxes NHWC bf16
__global__ void k_box2(const __bf16* __restrict__ t1, const float* __restrict__ w2,
                       const float* __restrict__ b2, __bf16* __restrict__ boxes) {
  int tid = blockIdx.x * 256 + threadIdx.x;            // 131072
  int m = tid >> 2, n = tid & 3;
  float s = b2[n];
  for (int c = 0; c < 64; ++c) s += (float)t1[(size_t)m * 64 + c] * w2[n * 64 + c];
  boxes[tid] = (__bf16)sigmoidf_(s);
}

// edge_net conv3x3 4->64 + bias -> t2 f32 [m][64]
__global__ void k_edge1(const __bf16* __restrict__ boxes, const float* __restrict__ w1,
                        const float* __restrict__ b1, float* __restrict__ t2) {
  int tid = blockIdx.x * 256 + threadIdx.x;            // 2097152
  int m = tid >> 6, co = tid & 63;
  int b = m >> 12, y = (m >> 6) & 63, xw = m & 63;
  float s = b1[co];
  for (int kh = 0; kh < 3; ++kh) {
    int yy = y + kh - 1; if ((unsigned)yy >= 64u) continue;
    for (int kw = 0; kw < 3; ++kw) {
      int xx = xw + kw - 1; if ((unsigned)xx >= 64u) continue;
      const __bf16* px = boxes + (size_t)((b * 64 + yy) * 64 + xx) * 4;
      for (int ci = 0; ci < 4; ++ci)
        s += (float)px[ci] * w1[((co * 4 + ci) * 3 + kh) * 3 + kw];
    }
  }
  t2[tid] = s;
}

// GroupNorm(8,64) stats per (b, group)
__global__ void k_gnstats(const float* __restrict__ t2, float* __restrict__ st) {
  __shared__ float ssum[256], ssq[256];
  int bg = blockIdx.x;                                  // 64
  int b = bg >> 3, g = bg & 7;
  float s = 0.f, q = 0.f;
  for (int idx = threadIdx.x; idx < 32768; idx += 256) {
    int sp = idx >> 3, c8 = idx & 7;
    float v = t2[(size_t)(b * 4096 + sp) * 64 + g * 8 + c8];
    s += v; q += v * v;
  }
  ssum[threadIdx.x] = s; ssq[threadIdx.x] = q;
  __syncthreads();
  for (int o = 128; o > 0; o >>= 1) {
    if (threadIdx.x < o) {
      ssum[threadIdx.x] += ssum[threadIdx.x + o];
      ssq[threadIdx.x]  += ssq[threadIdx.x + o];
    }
    __syncthreads();
  }
  if (threadIdx.x == 0) {
    float mean = ssum[0] * (1.f / 32768.f);
    st[bg * 2] = mean;
    st[bg * 2 + 1] = ssq[0] * (1.f / 32768.f) - mean * mean;
  }
}

// GN apply + GELU + conv1x1 64->8 -> edge f32 [b,h,y,x]
__global__ void k_edge2(const float* __restrict__ t2, const float* __restrict__ st,
                        const float* __restrict__ gg, const float* __restrict__ gb,
                        const float* __restrict__ w2, const float* __restrict__ b2,
                        float* __restrict__ edge) {
  int tid = blockIdx.x * 256 + threadIdx.x;            // 262144
  int m = tid >> 3, hh = tid & 7;
  int b = m >> 12, y = (m >> 6) & 63, xw = m & 63;
  float s = b2[hh];
  for (int c = 0; c < 64; ++c) {
    int g = c >> 3;
    float mean = st[(b * 8 + g) * 2], var = st[(b * 8 + g) * 2 + 1];
    float xn = (t2[(size_t)m * 64 + c] - mean) * rsqrtf(var + 1e-5f) * gg[c] + gb[c];
    s += gelu_exact(xn) * w2[hh * 64 + c];
  }
  edge[((size_t)(b * 8 + hh) * 64 + y) * 64 + xw] = s;
}

// ---------------------------------------------------------------------------
// qkv 1x1 conv (WMMA GEMM M=32768,N=1536,K=512); wave: 16 rows x 4 N-tiles.
//   q_t,k_t: [b,h,i, x, y]   (q scaled by HD^-0.5);  v_t: [b,h,i, y, x]
// ---------------------------------------------------------------------------
__global__ void k_qkv(const __bf16* __restrict__ xh, const __bf16* __restrict__ pw,
                      const float* __restrict__ bias,
                      __bf16* __restrict__ qt, __bf16* __restrict__ kt,
                      __bf16* __restrict__ vt) {
  int wave = threadIdx.x >> 5, lane = threadIdx.x & 31;
  int m0 = blockIdx.x * 16;
  int ntb = (blockIdx.y * 4 + wave) * 4;               // 96 N-tiles total
  int row = lane & 15;
  int m = m0 + row;
  int b = m >> 12, y = (m >> 6) & 63, xw = m & 63;
  int pix = (b * PW + y + 1) * PW + (xw + 1);          // interior of padded xh
  v8f acc[4] = {};
  for (int s = 0; s < 16; ++s) {
    v16bf a = load_frag_rm(xh, pix, 512, s * 32);
#pragma unroll
    for (int nt = 0; nt < 4; ++nt) {
      v16bf bf = load_frag_packed(pw + ((size_t)((s * 96 + ntb + nt) * 32 + lane) << 4));
      acc[nt] = __builtin_amdgcn_wmma_f32_16x16x32_bf16(false, a, false, bf,
                                                        (short)0, acc[nt], false, false);
    }
  }
  int col = lane & 15;
  int rb = (lane & 16) ? 8 : 0;
#pragma unroll
  for (int nt = 0; nt < 4; ++nt) {
    int n = (ntb + nt) * 16 + col;
    int part = n >> 9, hh = (n >> 6) & 7, i = n & 63;
    for (int r = 0; r < 8; ++r) {
      int mr = m0 + rb + r;
      int bb = mr >> 12, yr = (mr >> 6) & 63, xr = mr & 63;
      float v = acc[nt][r] + bias[n];
      size_t sl = ((size_t)(bb * 8 + hh) * 64 + i) * 4096;
      if (part == 0)      qt[sl + xr * 64 + yr] = (__bf16)(v * 0.125f);
      else if (part == 1) kt[sl + xr * 64 + yr] = (__bf16)v;
      else                vt[sl + yr * 64 + xr] = (__bf16)v;
    }
  }
}

// Scores per slice (b,h,i): S[w][W] = sum_y q[w][y] k_t[W][y]  (+ edge[b,h,i,w])
__global__ void k_scores(const __bf16* __restrict__ qt, const __bf16* __restrict__ kt,
                         const float* __restrict__ edge, float* __restrict__ S) {
  int wave = threadIdx.x >> 5, lane = threadIdx.x & 31;
  int slice = blockIdx.x;                               // 4096
  const __bf16* qs = qt + (size_t)slice * 4096;
  const __bf16* ks = kt + (size_t)slice * 4096;
  int bh = slice >> 6, i = slice & 63;
  int m0 = wave * 16;
  int row = lane & 15, col = lane & 15;
  int rb = (lane & 16) ? 8 : 0;
  v16bf a0 = load_frag_rm(qs, m0 + row, 64, 0);
  v16bf a1 = load_frag_rm(qs, m0 + row, 64, 32);
#pragma unroll
  for (int nt = 0; nt < 4; ++nt) {
    v8f acc = {};
    v16bf b0 = load_frag_rm(ks, nt * 16 + row, 64, 0);
    v16bf b1 = load_frag_rm(ks, nt * 16 + row, 64, 32);
    acc = __builtin_amdgcn_wmma_f32_16x16x32_bf16(false, a0, false, b0, (short)0, acc,
                                                  false, false);
    acc = __builtin_amdgcn_wmma_f32_16x16x32_bf16(false, a1, false, b1, (short)0, acc,
                                                  false, false);
    for (int r = 0; r < 8; ++r) {
      int wr = m0 + rb + r;
      S[(size_t)slice * 4096 + wr * 64 + nt * 16 + col] =
          acc[r] + edge[(bh * 64 + i) * 64 + wr];
    }
  }
}

// Softmax over i (stride 4096) -> attn bf16, same [bh,i,w,W] layout
__global__ void k_softmax(const float* __restrict__ S, __bf16* __restrict__ attn) {
  int tid = blockIdx.x * 256 + threadIdx.x;            // 262144 = (bh,w,W)
  int bh = tid >> 12;
  int w = (tid >> 6) & 63;
  int W = tid & 63;
  const float* p = S + (size_t)bh * 262144 + w * 64 + W;
  float mx = -1e30f;
  for (int i = 0; i < 64; ++i) mx = fmaxf(mx, p[i * 4096]);
  float sum = 0.f;
  for (int i = 0; i < 64; ++i) sum += expf(p[i * 4096] - mx);
  float inv = 1.0f / sum;
  __bf16* q = attn + (size_t)bh * 262144 + w * 64 + W;
  for (int i = 0; i < 64; ++i) q[i * 4096] = (__bf16)(expf(p[i * 4096] - mx) * inv);
}

// agg per slice: C[y][w] = sum_W v[y][W] attn[w][W]  -> padded NHWC bf16
__global__ void k_agg(const __bf16* __restrict__ vt, const __bf16* __restrict__ attn,
                      __bf16* __restrict__ aggh) {
  int wave = threadIdx.x >> 5, lane = threadIdx.x & 31;
  int slice = blockIdx.x;                               // 4096
  const __bf16* vs = vt + (size_t)slice * 4096;
  const __bf16* as = attn + (size_t)slice * 4096;
  int b = slice >> 9, ci = slice & 511;
  int m0 = wave * 16;
  int row = lane & 15, col = lane & 15;
  int rb = (lane & 16) ? 8 : 0;
  v16bf a0 = load_frag_rm(vs, m0 + row, 64, 0);
  v16bf a1 = load_frag_rm(vs, m0 + row, 64, 32);
#pragma unroll
  for (int nt = 0; nt < 4; ++nt) {
    v8f acc = {};
    v16bf b0 = load_frag_rm(as, nt * 16 + row, 64, 0);
    v16bf b1 = load_frag_rm(as, nt * 16 + row, 64, 32);
    acc = __builtin_amdgcn_wmma_f32_16x16x32_bf16(false, a0, false, b0, (short)0, acc,
                                                  false, false);
    acc = __builtin_amdgcn_wmma_f32_16x16x32_bf16(false, a1, false, b1, (short)0, acc,
                                                  false, false);
    for (int r = 0; r < 8; ++r) {
      int yr = m0 + rb + r;
      int wc = nt * 16 + col;
      aggh[((size_t)(b * PW + yr + 1) * PW + (wc + 1)) * 512 + ci] = (__bf16)acc[r];
    }
  }
}

// ---------------------------------------------------------------------------
// fusion conv3x3 (2C->C): M=32768, N=512, K=9216.  Wave: 16 rows x 4 N-tiles.
// Weights (nt-major pack) staged via TDM into double-buffered LDS.
// ---------------------------------------------------------------------------
#define FUS_KSTEPS 288

__global__ void k_fusion(const __bf16* __restrict__ xh, const __bf16* __restrict__ aggh,
                         const __bf16* __restrict__ pw, const float* __restrict__ bias,
                         float* __restrict__ out) {
  __shared__ __bf16 wlds[2][4][2048];                   // [buf][wave][4nt*512] = 32KB
  int wave = threadIdx.x >> 5, lane = threadIdx.x & 31;
  int m0 = blockIdx.x * 16;
  int ntb = (blockIdx.y * 4 + wave) * 4;                // 32 N-tiles total
  int row = lane & 15;
  int m = m0 + row;
  int b = m >> 12, y = (m >> 6) & 63, xw = m & 63;
  v8f acc[4] = {};
#if HAVE_TDM
  // low 32 bits of a flat LDS address == LDS byte offset (ISA aperture rule)
  unsigned ldsbase = (unsigned)(uintptr_t)&wlds[0][0][0];
  tdm_load_w(pw + (size_t)ntb * FUS_KSTEPS * 512, ldsbase + wave * 4096u);
#endif
  for (int s = 0; s < FUS_KSTEPS; ++s) {
    int tap = s >> 5;                                   // 1024 cin = 32 steps/tap
    int c0 = (s & 31) * 32;
    int pix = (b * PW + y + tap / 3) * PW + (xw + tap % 3);
    const __bf16* src = ((c0 < 512) ? xh : aggh) + (size_t)pix * 512 + (c0 & 511);
    FragB fa;
    int aoff = (lane & 16) ? 8 : 0;
    fa.u[0] = *(const uint4*)(src + aoff);
    fa.u[1] = *(const uint4*)(src + aoff + 16);
    v16bf a = fa.v;
#if HAVE_TDM
    if (s + 1 < FUS_KSTEPS) {
      asm volatile("s_wait_dscnt 0x0" ::: "memory");    // buffer reuse fence
      tdm_load_w(pw + ((size_t)ntb * FUS_KSTEPS + (s + 1)) * 512,
                 ldsbase + (((s + 1) & 1) * 4 + wave) * 4096u);
      WAIT_TENSORCNT(1);                                // step s staged
    } else {
      WAIT_TENSORCNT(0);
    }
    unsigned bufoff = ldsbase + ((s & 1) * 4 + wave) * 4096u + lane * 32u;
#endif
#pragma unroll
    for (int nt = 0; nt < 4; ++nt) {
#if HAVE_TDM
      v16bf bf = load_frag_lds(bufoff + nt * 1024u);    // ds_load_b128 x2
#else
      v16bf bf = load_frag_packed(pw + ((size_t)(ntb + nt) * FUS_KSTEPS + s) * 512 +
                                  lane * 16);
#endif
      acc[nt] = __builtin_amdgcn_wmma_f32_16x16x32_bf16(false, a, false, bf,
                                                        (short)0, acc[nt], false, false);
    }
  }
  int col = lane & 15;
  int rb = (lane & 16) ? 8 : 0;
#pragma unroll
  for (int nt = 0; nt < 4; ++nt) {
    int n = (ntb + nt) * 16 + col;
    for (int r = 0; r < 8; ++r) {
      int mr = m0 + rb + r;
      int bb = mr >> 12, yr = (mr >> 6) & 63, xr = mr & 63;
      out[((size_t)(bb * 512 + n) * 64 + yr) * 64 + xr] = acc[nt][r] + bias[n];
    }
  }
}

// BatchNorm stats per channel over (B,H,W)
__global__ void k_bnstats(const float* __restrict__ out, float* __restrict__ st) {
  __shared__ float ssum[256], ssq[256];
  int c = blockIdx.x;                                   // 512
  float s = 0.f, q = 0.f;
  for (int b = 0; b < 8; ++b)
    for (int sp = threadIdx.x; sp < 4096; sp += 256) {
      float v = out[(size_t)(b * 512 + c) * 4096 + sp];
      s += v; q += v * v;
    }
  ssum[threadIdx.x] = s; ssq[threadIdx.x] = q;
  __syncthreads();
  for (int o = 128; o > 0; o >>= 1) {
    if (threadIdx.x < o) {
      ssum[threadIdx.x] += ssum[threadIdx.x + o];
      ssq[threadIdx.x]  += ssq[threadIdx.x + o];
    }
    __syncthreads();
  }
  if (threadIdx.x == 0) {
    float mean = ssum[0] * (1.f / 32768.f);
    st[c] = mean;
    st[512 + c] = ssq[0] * (1.f / 32768.f) - mean * mean;
  }
}

// BN apply + SiLU (in place on d_out)
__global__ void k_bnsilu(float* __restrict__ out, const float* __restrict__ st,
                         const float* __restrict__ g, const float* __restrict__ bb) {
  int tid = blockIdx.x * 256 + threadIdx.x;            // 16777216
  int c = (tid >> 12) & 511;
  float v = out[tid];
  float yn = (v - st[c]) * rsqrtf(st[512 + c] + 1e-5f) * g[c] + bb[c];
  out[tid] = yn * sigmoidf_(yn);
}

// ---------------------------------------------------------------------------

extern "C" void kernel_launch(void* const* d_in, const int* in_sizes, int n_in,
                              void* d_out, int out_size, void* d_ws, size_t ws_size,
                              hipStream_t stream) {
  const float* x    = (const float*)d_in[0];
  const float* bw1  = (const float*)d_in[1];
  const float* bb1  = (const float*)d_in[2];
  const float* bw2  = (const float*)d_in[3];
  const float* bb2  = (const float*)d_in[4];
  const float* ew1  = (const float*)d_in[5];
  const float* eb1  = (const float*)d_in[6];
  const float* gng  = (const float*)d_in[7];
  const float* gnb  = (const float*)d_in[8];
  const float* ew2  = (const float*)d_in[9];
  const float* eb2  = (const float*)d_in[10];
  const float* qkvw = (const float*)d_in[11];
  const float* qkvb = (const float*)d_in[12];
  const float* fw   = (const float*)d_in[13];
  const float* fb   = (const float*)d_in[14];
  const float* bng  = (const float*)d_in[15];
  const float* bnb  = (const float*)d_in[16];
  float* out = (float*)d_out;

  char* ws = (char*)d_ws;
  size_t off = 0;
  auto carve = [&](size_t bytes) -> void* {
    void* p = ws + off;
    off += (bytes + 255) & ~(size_t)255;
    return p;
  };

  const size_t NE = 16777216;                           // B*C*H*W
  __bf16* xh    = (__bf16*)carve((size_t)PAD_ELEMS * 2);
  __bf16* aggh  = (__bf16*)carve((size_t)PAD_ELEMS * 2);
  __bf16* qt    = (__bf16*)carve(NE * 2);
  __bf16* kt    = (__bf16*)carve(NE * 2);
  __bf16* vt    = (__bf16*)carve(NE * 2);
  float*  S     = (float*)carve(NE * 4);
  __bf16* attn  = (__bf16*)carve(NE * 2);
  __bf16* t1    = (__bf16*)carve((size_t)32768 * 64 * 2);
  __bf16* boxes = (__bf16*)carve((size_t)32768 * 4 * 2);
  float*  t2    = (float*)carve((size_t)32768 * 64 * 4);
  float*  edge  = (float*)carve((size_t)262144 * 4);
  float*  gnst  = (float*)carve(128 * 4);
  float*  bnst  = (float*)carve(1024 * 4);
  __bf16* pwbox = (__bf16*)carve((size_t)4608 * 64 * 2);
  __bf16* pwqkv = (__bf16*)carve((size_t)512 * 1536 * 2);
  __bf16* pwfus = (__bf16*)carve((size_t)9216 * 512 * 2);

  // prep: padded cast, agg halo zero, weight packs
  k_cast_x_pad<<<PAD_ELEMS / 256, 256, 0, stream>>>(x, xh);
  k_zero_bf16<<<PAD_ELEMS / 256, 256, 0, stream>>>(aggh);
  k_pack_w<<<1152, 256, 0, stream>>>(bw1, pwbox, 4608, 64, 512, 1, 0);
  k_pack_w<<<3072, 256, 0, stream>>>(qkvw, pwqkv, 512, 1536, 512, 0, 0);
  k_pack_w<<<18432, 256, 0, stream>>>(fw, pwfus, 9216, 512, 1024, 1, 1);  // nt-major

  // box_net + edge_net
  k_box_conv1<<<512, 128, 0, stream>>>(xh, pwbox, bb1, t1);
  k_box2<<<512, 256, 0, stream>>>(t1, bw2, bb2, boxes);
  k_edge1<<<8192, 256, 0, stream>>>(boxes, ew1, eb1, t2);
  k_gnstats<<<64, 256, 0, stream>>>(t2, gnst);
  k_edge2<<<1024, 256, 0, stream>>>(t2, gnst, gng, gnb, ew2, eb2, edge);

  // attention
  k_qkv<<<dim3(2048, 6), 128, 0, stream>>>(xh, pwqkv, qkvb, qt, kt, vt);
  k_scores<<<4096, 128, 0, stream>>>(qt, kt, edge, S);
  k_softmax<<<1024, 256, 0, stream>>>(S, attn);
  k_agg<<<4096, 128, 0, stream>>>(vt, attn, aggh);

  // fusion + BN + SiLU
  k_fusion<<<dim3(2048, 2), 128, 0, stream>>>(xh, aggh, pwfus, fb, out);
  k_bnstats<<<512, 256, 0, stream>>>(out, bnst);
  k_bnsilu<<<65536, 256, 0, stream>>>(out, bnst, bng, bnb);
}
